// SSL_90031104458999
// MI455X (gfx1250) — matrix operand order repaired and checked by reading, not compiled, not run on hardware
//
#include <hip/hip_runtime.h>

// ---------------------------------------------------------------------------
// LRU block for MI455X (gfx1250): bf16 WMMA GEMMs (async-LDS staged, double
// buffered) + chunked complex scan.
// ---------------------------------------------------------------------------

#define AS1 __attribute__((address_space(1)))
#define AS3 __attribute__((address_space(3)))

#if __has_builtin(__builtin_amdgcn_global_load_async_to_lds_b128)
#define HAVE_ASYNC 1
#else
#define HAVE_ASYNC 0
#endif

typedef __attribute__((ext_vector_type(4)))  __bf16 v4bf;
typedef __attribute__((ext_vector_type(8)))  __bf16 v8bf;
typedef __attribute__((ext_vector_type(16))) __bf16 v16bf;
typedef __attribute__((ext_vector_type(8)))  float  v8f;
typedef __attribute__((ext_vector_type(4)))  float  v4f;
typedef __attribute__((ext_vector_type(4)))  int    v4i;

template <class X, class Y> struct same_t { static constexpr bool v = false; };
template <class X> struct same_t<X, X>    { static constexpr bool v = true;  };

// Problem sizes (fixed by the reference)
static constexpr int B_   = 4;
static constexpr int L_   = 4096;
static constexpr int DM   = 1024;   // d_model
static constexpr int DS   = 256;    // d_state
static constexpr int DH   = 4096;   // d_hidden
static constexpr int MT   = B_ * L_;          // 16384 token rows
static constexpr int TCH  = 128;              // scan chunk length
static constexpr int NCH  = L_ / TCH;         // 32 chunks per sequence

// GEMM tiling
static constexpr int BM = 128, BN = 128, BK = 64;
static constexpr int LDS_PITCH = BK + 8;      // 72 bf16 = 144B (16B aligned, conflict-free)

enum { EPI_F32 = 0, EPI_BF16 = 1, EPI_GELU = 2, EPI_RESID = 3 };

// ---------------------------------------------------------------------------
// Address-space helpers (integer round-trip: always legal, bit-identical for
// global memory; low 32 bits of a generic LDS pointer are the LDS address).
// ---------------------------------------------------------------------------
template <typename T>
__device__ __forceinline__ const AS1 T* gc(const T* p) {
    return (const AS1 T*)(unsigned long long)p;
}
template <typename T>
__device__ __forceinline__ AS1 T* gcm(T* p) {
    return (AS1 T*)(unsigned long long)p;
}
__device__ __forceinline__ AS3 v4i* lc4(void* p) {
    return (AS3 v4i*)(unsigned)(unsigned long long)p;
}
template <typename T>
__device__ __forceinline__ AS1 v4i* g4(const AS1 T* p) {
    return (AS1 v4i*)(unsigned long long)p;
}

template <int N>
__device__ __forceinline__ void wait_async() {
#if HAVE_ASYNC
# if __has_builtin(__builtin_amdgcn_s_wait_asynccnt)
    __builtin_amdgcn_s_wait_asynccnt(N);
# else
    asm volatile("s_wait_asynccnt %0" :: "i"(N) : "memory");
# endif
#endif
}

// ---------------------------------------------------------------------------
// Small helpers
// ---------------------------------------------------------------------------
__device__ __forceinline__ v8bf load8(const AS1 __bf16* p) {
    return *(const AS1 v8bf*)p;
}
__device__ __forceinline__ v8bf load8(const AS1 float* p) {
    v4f a = *(const AS1 v4f*)p;
    v4f b = *(const AS1 v4f*)(p + 4);
    v8bf r;
    r[0] = (__bf16)a[0]; r[1] = (__bf16)a[1]; r[2] = (__bf16)a[2]; r[3] = (__bf16)a[3];
    r[4] = (__bf16)b[0]; r[5] = (__bf16)b[1]; r[6] = (__bf16)b[2]; r[7] = (__bf16)b[3];
    return r;
}
__device__ __forceinline__ v16bf cat16(v8bf a, v8bf b) {
    return __builtin_shufflevector(a, b, 0,1,2,3,4,5,6,7,8,9,10,11,12,13,14,15);
}
__device__ __forceinline__ float gelu_tanh(float v) {
    const float c0 = 0.7978845608028654f;   // sqrt(2/pi)
    const float c1 = 0.044715f;
    float t = tanhf(c0 * (v + c1 * v * v * v));
    return 0.5f * v * (1.0f + t);
}

// ---------------------------------------------------------------------------
// Parameter precompute: lam, gamma, lam^TCH, sigmoid(res_logit)
// P layout (floats): [0]=sig, [8..]=lam_re, [8+256..]=lam_im,
// [8+512..]=lamT_re, [8+768..]=lamT_im, [8+1024..]=gamma
// ---------------------------------------------------------------------------
__global__ __launch_bounds__(256) void k_params(const float* nu_log,
                                                const float* theta_log,
                                                const float* res_logit,
                                                float* P) {
    int n = threadIdx.x;
    double nu = exp((double)nu_log[n]);
    double th = exp((double)theta_log[n]);
    double r  = exp(-nu);
    P[8 + 0*DS + n] = (float)(r * cos(th));
    P[8 + 1*DS + n] = (float)(r * sin(th));
    double rT = exp(-(double)TCH * nu);
    double aT = (double)TCH * th;
    P[8 + 2*DS + n] = (float)(rT * cos(aT));
    P[8 + 3*DS + n] = (float)(rT * sin(aT));
    P[8 + 4*DS + n] = (float)sqrt(1.0 - r * r);
    if (n == 0) P[0] = 1.0f / (1.0f + expf(-res_logit[0]));
}

// ---------------------------------------------------------------------------
// fp32 -> bf16 conversion (8-element chunks, grid-stride)
// ---------------------------------------------------------------------------
__global__ __launch_bounds__(256) void k_f32_to_bf16(const float* __restrict__ src_,
                                                     __bf16* __restrict__ dst_,
                                                     int nchunks8) {
    const AS1 float* src = gc(src_);
    AS1 __bf16* dst = gcm(dst_);
    int stride = gridDim.x * blockDim.x;
    for (int c = blockIdx.x * blockDim.x + threadIdx.x; c < nchunks8; c += stride) {
        v8bf v = load8(src + (long long)c * 8);
        *(AS1 v8bf*)(dst + (long long)c * 8) = v;
    }
}

// Bn = gamma[n] * (B_re + i B_im), bf16 planes.  idx over DS*DM elements.
__global__ __launch_bounds__(256) void k_make_bn(const float* __restrict__ Bre_,
                                                 const float* __restrict__ Bim_,
                                                 const float* __restrict__ P_,
                                                 __bf16* __restrict__ bn_re_,
                                                 __bf16* __restrict__ bn_im_) {
    const AS1 float* Bre = gc(Bre_);
    const AS1 float* Bim = gc(Bim_);
    const AS1 float* P   = gc(P_);
    AS1 __bf16* bn_re = gcm(bn_re_);
    AS1 __bf16* bn_im = gcm(bn_im_);
    int idx = blockIdx.x * blockDim.x + threadIdx.x;   // < DS*DM = 262144
    int n = idx >> 10;                                  // row (state index)
    float g = P[8 + 4*DS + n];
    bn_re[idx] = (__bf16)(g * Bre[idx]);
    bn_im[idx] = (__bf16)(g * Bim[idx]);
}

// ---------------------------------------------------------------------------
// Chunked complex scan.  Bu layout: [(b*L + l)*DS + n], fp32 planes.
// Pass 1: in-place local scan per (b, n, chunk); emit chunk-end local state.
// ---------------------------------------------------------------------------
__global__ __launch_bounds__(256) void k_scan_local(float* __restrict__ bu_re_,
                                                    float* __restrict__ bu_im_,
                                                    const float* __restrict__ P_,
                                                    float* __restrict__ cl_re_,
                                                    float* __restrict__ cl_im_) {
    AS1 float* bu_re = gcm(bu_re_);
    AS1 float* bu_im = gcm(bu_im_);
    const AS1 float* P = gc(P_);
    int n = threadIdx.x;            // 0..255
    int c = blockIdx.x;             // chunk 0..31
    int b = blockIdx.y;             // batch 0..3
    float lre = P[8 + 0*DS + n], lim = P[8 + 1*DS + n];
    long long base = ((long long)b * L_ + (long long)c * TCH) * DS + n;
    float sre = 0.f, sim = 0.f;
    for (int i = 0; i < TCH; ++i) {
        long long idx = base + (long long)i * DS;
        float br = bu_re[idx], bi = bu_im[idx];
        float nre = lre * sre - lim * sim + br;
        float nim = lre * sim + lim * sre + bi;
        sre = nre; sim = nim;
        bu_re[idx] = sre; bu_im[idx] = sim;
    }
    int ch = (b * DS + n) * NCH + c;
    gcm(cl_re_)[ch] = sre; gcm(cl_im_)[ch] = sim;
}

// Pass 2: propagate carries across chunks with lam^TCH (1024 channels).
__global__ __launch_bounds__(256) void k_scan_carry(const float* __restrict__ P_,
                                                    const float* __restrict__ cl_re_,
                                                    const float* __restrict__ cl_im_,
                                                    float* __restrict__ ci_re_,
                                                    float* __restrict__ ci_im_) {
    const AS1 float* P = gc(P_);
    const AS1 float* cl_re = gc(cl_re_);
    const AS1 float* cl_im = gc(cl_im_);
    AS1 float* ci_re = gcm(ci_re_);
    AS1 float* ci_im = gcm(ci_im_);
    int ch = blockIdx.x * blockDim.x + threadIdx.x;   // 0..1023 = b*DS + n
    int n = ch & (DS - 1);
    float lTre = P[8 + 2*DS + n], lTim = P[8 + 3*DS + n];
    float sre = 0.f, sim = 0.f;                       // S_{-1} = 0
    for (int c = 0; c < NCH; ++c) {
        int k = ch * NCH + c;
        ci_re[k] = sre; ci_im[k] = sim;               // carry INTO chunk c
        float nre = lTre * sre - lTim * sim + cl_re[k];
        float nim = lTre * sim + lTim * sre + cl_im[k];
        sre = nre; sim = nim;
    }
}

// Pass 3: fix up full states; write bf16 planes (im plane negated for the
// y-projection GEMM) and the final state output.
__global__ __launch_bounds__(256) void k_scan_fix(const float* __restrict__ bu_re_,
                                                  const float* __restrict__ bu_im_,
                                                  const float* __restrict__ P_,
                                                  const float* __restrict__ ci_re_,
                                                  const float* __restrict__ ci_im_,
                                                  __bf16* __restrict__ st_re_,
                                                  __bf16* __restrict__ st_imneg_,
                                                  float* __restrict__ st_out_) {
    const AS1 float* bu_re = gc(bu_re_);
    const AS1 float* bu_im = gc(bu_im_);
    const AS1 float* P = gc(P_);
    AS1 __bf16* st_re = gcm(st_re_);
    AS1 __bf16* st_imneg = gcm(st_imneg_);
    int n = threadIdx.x;
    int c = blockIdx.x;
    int b = blockIdx.y;
    float lre = P[8 + 0*DS + n], lim = P[8 + 1*DS + n];
    int ch = (b * DS + n) * NCH + c;
    float cre = gc(ci_re_)[ch], cim = gc(ci_im_)[ch];
    // p = lam^(i+1) * carry, start at lam * carry
    float pre = lre * cre - lim * cim;
    float pim = lre * cim + lim * cre;
    long long base = ((long long)b * L_ + (long long)c * TCH) * DS + n;
    float sre = 0.f, sim = 0.f;
    for (int i = 0; i < TCH; ++i) {
        long long idx = base + (long long)i * DS;
        sre = bu_re[idx] + pre;
        sim = bu_im[idx] + pim;
        st_re[idx]    = (__bf16)sre;
        st_imneg[idx] = (__bf16)(-sim);
        float npre = lre * pre - lim * pim;
        float npim = lre * pim + lim * pre;
        pre = npre; pim = npim;
    }
    if (c == NCH - 1) {                                // final state (b, n)
        gcm(st_out_)[(b * DS + n) * 2 + 0] = sre;
        gcm(st_out_)[(b * DS + n) * 2 + 1] = sim;
    }
}

// ---------------------------------------------------------------------------
// Generic bf16 WMMA GEMM.  D(M,N) = sum_p A_p(M,K_p) * B_p^T  (+ epilogue)
//   KMAJOR=false: B source row-major [N][K] (already B^T)
//   KMAJOR=true : B source row-major [K][N] (transposed while staging)
//   BSRC float -> converted to bf16 on the fly (sync); bf16 -> async-LDS.
// Block: 256 threads = 8 waves (wave32), 128x128 tile, BK=64, double buffer,
// software pipelined with GLOBAL_LOAD_ASYNC_TO_LDS (ASYNCcnt).
// ---------------------------------------------------------------------------
struct GemmArgs {
    const __bf16* A0; const __bf16* A1; const __bf16* A2;
    const void*   B0; const void*   B1; const void*   B2;
    int K0, K1, K2;
    int np;
    int M, N;
    float*  outF;
    __bf16* outBf;
    const float* resid;   // x (fp32) for residual epilogue
    const float* P;       // params (sig at P[0])
};

template <bool KMAJOR, typename BSRC, int EPI>
__global__ __launch_bounds__(256) void k_gemm(GemmArgs g) {
    constexpr bool BBF = same_t<BSRC, __bf16>::v;
    // async ops issued per stage() per wave: 4 for A, +4 for B when async-able
    constexpr int ASYNC_N = HAVE_ASYNC ? ((!KMAJOR && BBF) ? 8 : 4) : 0;

    __shared__ __bf16 sA[2][BM * LDS_PITCH];
    __shared__ __bf16 sB[2][BN * LDS_PITCH];

    const int tid  = threadIdx.x;
    const int wid  = tid >> 5;
    const int lane = tid & 31;
    const int lr   = lane & 15;
    const int hi   = lane >> 4;
    const int wm   = (wid >> 1) * 32;    // wave row offset in tile
    const int wn   = (wid & 1) * 64;     // wave col offset in tile
    const int m0   = blockIdx.y * BM;
    const int n0   = blockIdx.x * BN;

    v8f acc[2][4];
    v8f zero;
    #pragma unroll
    for (int e = 0; e < 8; ++e) zero[e] = 0.0f;
    #pragma unroll
    for (int im = 0; im < 2; ++im)
        #pragma unroll
        for (int in_ = 0; in_ < 4; ++in_) acc[im][in_] = zero;

    const __bf16* As_[3] = { g.A0, g.A1, g.A2 };
    const BSRC*   Bs_[3] = { (const BSRC*)g.B0, (const BSRC*)g.B1, (const BSRC*)g.B2 };
    const int     Ks_[3] = { g.K0, g.K1, g.K2 };
    const int nt0 = g.K0 / BK;
    const int nt1 = g.K1 / BK;
    const int nt2 = g.K2 / BK;
    const int total = nt0 + nt1 + nt2;

    auto stage = [&](int t, int buf) {
        int p, k0;
        if (t < nt0)            { p = 0; k0 = t * BK; }
        else if (t < nt0 + nt1) { p = 1; k0 = (t - nt0) * BK; }
        else                    { p = 2; k0 = (t - nt0 - nt1) * BK; }
        const int K = Ks_[p];
        const AS1 __bf16* A = gc(As_[p]);
        __bf16* dA = &sA[buf][0];
        __bf16* dB = &sB[buf][0];

        // ---- A tile (BM x BK bf16): 4 x b128 per thread ----
        #pragma unroll
        for (int j = 0; j < 4; ++j) {
            int ci = tid + j * 256, row = ci >> 3, col0 = (ci & 7) * 8;
            const AS1 __bf16* src = A + (long long)(m0 + row) * K + k0 + col0;
            __bf16* dst = dA + row * LDS_PITCH + col0;
#if HAVE_ASYNC
            __builtin_amdgcn_global_load_async_to_lds_b128(g4(src), lc4(dst), 0, 0);
#else
            *(v8bf*)dst = load8(src);
#endif
        }
        // ---- B tile ----
        if constexpr (!KMAJOR) {
            if constexpr (BBF) {
                const AS1 __bf16* Bp = gc((const __bf16*)Bs_[p]);
                #pragma unroll
                for (int j = 0; j < 4; ++j) {
                    int ci = tid + j * 256, row = ci >> 3, col0 = (ci & 7) * 8;
                    const AS1 __bf16* src = Bp + (long long)(n0 + row) * K + k0 + col0;
                    __bf16* dst = dB + row * LDS_PITCH + col0;
#if HAVE_ASYNC
                    __builtin_amdgcn_global_load_async_to_lds_b128(g4(src), lc4(dst), 0, 0);
#else
                    *(v8bf*)dst = load8(src);
#endif
                }
            } else {
                const AS1 float* Bp = gc((const float*)Bs_[p]);
                v8bf tmp[4];
                #pragma unroll
                for (int j = 0; j < 4; ++j) {
                    int ci = tid + j * 256, row = ci >> 3, col0 = (ci & 7) * 8;
                    tmp[j] = load8(Bp + (long long)(n0 + row) * K + k0 + col0);
                }
                #pragma unroll
                for (int j = 0; j < 4; ++j) {
                    int ci = tid + j * 256, row = ci >> 3, col0 = (ci & 7) * 8;
                    *(v8bf*)(dB + row * LDS_PITCH + col0) = tmp[j];
                }
            }
        } else {
            const AS1 BSRC* Bp = gc(Bs_[p]);
            v8bf tmp[4];
            #pragma unroll
            for (int j = 0; j < 4; ++j) {
                int ci = tid + j * 256, kr = ci >> 4, nc0 = (ci & 15) * 8;
                tmp[j] = load8(Bp + (long long)(k0 + kr) * g.N + n0 + nc0);
            }
            #pragma unroll
            for (int j = 0; j < 4; ++j) {
                int ci = tid + j * 256, kr = ci >> 4, nc0 = (ci & 15) * 8;
                #pragma unroll
                for (int e = 0; e < 8; ++e)
                    dB[(nc0 + e) * LDS_PITCH + kr] = tmp[j][e];
            }
        }
    };

    auto compute = [&](int buf) {
        const __bf16* pA = &sA[buf][0];
        const __bf16* pB = &sB[buf][0];
        #pragma unroll
        for (int kk = 0; kk < 2; ++kk) {
            const int kb = kk * 32;
            v16bf afr[2], bfr[4];
            #pragma unroll
            for (int im = 0; im < 2; ++im) {
                int m   = wm + im * 16 + lr;
                int off = m * LDS_PITCH + kb + hi * 8;
                v8bf p0 = *(const v8bf*)(pA + off);        // K = kA .. kA+7
                v8bf p1 = *(const v8bf*)(pA + off + 16);   // K = kA+16 .. kA+23
                afr[im] = cat16(p0, p1);
            }
            #pragma unroll
            for (int in_ = 0; in_ < 4; ++in_) {
                int n   = wn + in_ * 16 + lr;
                int off = n * LDS_PITCH + kb + hi * 8;
                v8bf p0 = *(const v8bf*)(pB + off);        // K = kB .. kB+7
                v8bf p1 = *(const v8bf*)(pB + off + 16);   // K = kB+16 .. kB+23
                bfr[in_] = cat16(p0, p1);
            }
            #pragma unroll
            for (int im = 0; im < 2; ++im)
                #pragma unroll
                for (int in_ = 0; in_ < 4; ++in_)
                    acc[im][in_] = __builtin_amdgcn_wmma_f32_16x16x32_bf16(
                        false, afr[im], false, bfr[in_],
                        0, acc[im][in_], false, false);
        }
    };

    // Software pipeline: stage t+1 async while computing t.
    stage(0, 0);
    for (int t = 0; t < total; ++t) {
        const int cur = t & 1;
        if (t + 1 < total) {
            stage(t + 1, cur ^ 1);
            wait_async<ASYNC_N>();   // all ops older than the just-issued stage done
        } else {
            wait_async<0>();
        }
        __syncthreads();             // buf[cur] visible to all waves
        compute(cur);
        __syncthreads();             // everyone done reading buf[cur]
    }

    // ---- epilogue ----
    float sig = 0.0f;
    if constexpr (EPI == EPI_RESID) sig = gc(g.P)[0];

    #pragma unroll
    for (int im = 0; im < 2; ++im) {
        #pragma unroll
        for (int in_ = 0; in_ < 4; ++in_) {
            #pragma unroll
            for (int r = 0; r < 8; ++r) {
                int row = m0 + wm + im * 16 + hi * 8 + r;   // M = r + 8*hi
                int col = n0 + wn + in_ * 16 + lr;          // N = lane&15
                long long idx = (long long)row * g.N + col;
                float v = acc[im][in_][r];
                if constexpr (EPI == EPI_F32) {
                    gcm(g.outF)[idx] = v;
                } else if constexpr (EPI == EPI_BF16) {
                    gcm(g.outBf)[idx] = (__bf16)v;
                } else if constexpr (EPI == EPI_GELU) {
                    gcm(g.outBf)[idx] = (__bf16)gelu_tanh(v);
                } else { // EPI_RESID
                    gcm(g.outF)[idx] = gc(g.resid)[idx] + sig * v;
                }
            }
        }
    }
}

// ---------------------------------------------------------------------------
// Host-side launcher
// ---------------------------------------------------------------------------
extern "C" void kernel_launch(void* const* d_in, const int* in_sizes, int n_in,
                              void* d_out, int out_size, void* d_ws, size_t ws_size,
                              hipStream_t stream) {
    (void)in_sizes; (void)n_in; (void)out_size; (void)ws_size;

    const float* x      = (const float*)d_in[0];   // (B,L,DM)
    const float* nu_log = (const float*)d_in[1];   // (DS)
    const float* th_log = (const float*)d_in[2];   // (DS)
    const float* B_re   = (const float*)d_in[3];   // (DS,DM)
    const float* B_im   = (const float*)d_in[4];   // (DS,DM)
    const float* C_re   = (const float*)d_in[5];   // (DM,DS)
    const float* C_im   = (const float*)d_in[6];   // (DM,DS)
    const float* D_mat  = (const float*)d_in[7];   // (DM,DM)
    const float* W1     = (const float*)d_in[8];   // (DM,DH)
    const float* W2     = (const float*)d_in[9];   // (DH,DM)
    const float* res_l  = (const float*)d_in[10];  // scalar

    float* out = (float*)d_out;                    // (B,L,DM) fp32
    float* st_out = out + (long long)MT * DM;      // (B,DS) complex64 (re,im pairs)

    // ---- workspace carve (all 256B aligned) ----
    char* w = (char*)d_ws;
    size_t off = 0;
    auto carve = [&](size_t bytes) {
        void* p = w + off;
        off += (bytes + 255) & ~(size_t)255;
        return p;
    };
    float*  P       = (float*) carve(8192);
    __bf16* x_bf    = (__bf16*)carve((size_t)MT * DM * 2);
    __bf16* bn_re   = (__bf16*)carve((size_t)DS * DM * 2);
    __bf16* bn_im   = (__bf16*)carve((size_t)DS * DM * 2);
    float*  bu_re   = (float*) carve((size_t)MT * DS * 4);
    float*  bu_im   = (float*) carve((size_t)MT * DS * 4);
    float*  cl_re   = (float*) carve((size_t)B_ * DS * NCH * 4);
    float*  cl_im   = (float*) carve((size_t)B_ * DS * NCH * 4);
    float*  ci_re   = (float*) carve((size_t)B_ * DS * NCH * 4);
    float*  ci_im   = (float*) carve((size_t)B_ * DS * NCH * 4);
    __bf16* st_re   = (__bf16*)carve((size_t)MT * DS * 2);
    __bf16* st_imn  = (__bf16*)carve((size_t)MT * DS * 2);
    __bf16* y_bf    = (__bf16*)carve((size_t)MT * DM * 2);
    __bf16* h_bf    = (__bf16*)carve((size_t)MT * DH * 2);

    // 1) parameters
    k_params<<<1, 256, 0, stream>>>(nu_log, th_log, res_l, P);

    // 2) x -> bf16 ; Bn = gamma * (B_re, B_im) -> bf16
    k_f32_to_bf16<<<2048, 256, 0, stream>>>(x, x_bf, MT * DM / 8);
    k_make_bn<<<(DS * DM) / 256, 256, 0, stream>>>(B_re, B_im, P, bn_re, bn_im);

    // 3) Bu = x @ Bn^T  (two real GEMMs, M=16384 N=256 K=1024)
    {
        GemmArgs a{};
        a.A0 = x_bf; a.B0 = bn_re; a.K0 = DM; a.np = 1;
        a.M = MT; a.N = DS; a.outF = bu_re; a.P = P;
        k_gemm<false, __bf16, EPI_F32><<<dim3(DS / BN, MT / BM), 256, 0, stream>>>(a);
        a.B0 = bn_im; a.outF = bu_im;
        k_gemm<false, __bf16, EPI_F32><<<dim3(DS / BN, MT / BM), 256, 0, stream>>>(a);
    }

    // 4) chunked complex scan
    k_scan_local<<<dim3(NCH, B_), 256, 0, stream>>>(bu_re, bu_im, P, cl_re, cl_im);
    k_scan_carry<<<(B_ * DS) / 256, 256, 0, stream>>>(P, cl_re, cl_im, ci_re, ci_im);
    k_scan_fix<<<dim3(NCH, B_), 256, 0, stream>>>(bu_re, bu_im, P, ci_re, ci_im,
                                                  st_re, st_imn, st_out);

    // 5) y = s_re@C_re^T + (-s_im)@C_im^T + x@D^T   (fused 3-pair GEMM, N=1024)
    {
        GemmArgs a{};
        a.A0 = st_re;  a.B0 = C_re;  a.K0 = DS;
        a.A1 = st_imn; a.B1 = C_im;  a.K1 = DS;
        a.A2 = x_bf;   a.B2 = D_mat; a.K2 = DM;
        a.np = 3; a.M = MT; a.N = DM; a.outBf = y_bf; a.P = P;
        k_gemm<false, float, EPI_BF16><<<dim3(DM / BN, MT / BM), 256, 0, stream>>>(a);
    }

    // 6) h = gelu(y @ W1)   (K-major B, M=16384 N=4096 K=1024)
    {
        GemmArgs a{};
        a.A0 = y_bf; a.B0 = W1; a.K0 = DM; a.np = 1;
        a.M = MT; a.N = DH; a.outBf = h_bf; a.P = P;
        k_gemm<true, float, EPI_GELU><<<dim3(DH / BN, MT / BM), 256, 0, stream>>>(a);
    }

    // 7) out = x + sig * (h @ W2)   (K-major B, M=16384 N=1024 K=4096)
    {
        GemmArgs a{};
        a.A0 = h_bf; a.B0 = W2; a.K0 = DH; a.np = 1;
        a.M = MT; a.N = DM; a.outF = out; a.resid = x; a.P = P;
        k_gemm<true, float, EPI_RESID><<<dim3(DM / BN, MT / BM), 256, 0, stream>>>(a);
    }
}
